// Content_SA_18717467476189
// MI455X (gfx1250) — matrix operand order repaired and checked by reading, not compiled, not run on hardware
//
#include <hip/hip_runtime.h>
#include <hip/hip_bf16.h>

typedef __attribute__((ext_vector_type(16))) _Float16 v16h;
typedef __attribute__((ext_vector_type(8)))  float    v8f;

#define HW_ 4096
#define C_  512

union FragH { v16h v; uint4 q[2]; };

// ---------------------------------------------------------------------------
// Per-channel mean / inv-std over 4096 spatial elems.  One block per channel.
// stats[c] = (mean, rsqrt(var+eps))
// ---------------------------------------------------------------------------
__global__ __launch_bounds__(256)
void meanvar_kernel(const float* __restrict__ x, float2* __restrict__ stats) {
  __shared__ float s1[256];
  __shared__ float s2[256];
  const int c = blockIdx.x;
  const float* row = x + (size_t)c * HW_;
  float sum = 0.f, ss = 0.f;
#pragma unroll
  for (int q = 0; q < 16; ++q) {
    float v = row[threadIdx.x + q * 256];
    sum += v; ss += v * v;
  }
  s1[threadIdx.x] = sum; s2[threadIdx.x] = ss;
  __syncthreads();
  for (int st = 128; st > 0; st >>= 1) {
    if (threadIdx.x < st) {
      s1[threadIdx.x] += s1[threadIdx.x + st];
      s2[threadIdx.x] += s2[threadIdx.x + st];
    }
    __syncthreads();
  }
  if (threadIdx.x == 0) {
    const float mean = s1[0] * (1.0f / HW_);
    const float var  = s2[0] * (1.0f / HW_) - mean * mean;
    stats[c] = make_float2(mean, rsqrtf(var + 1e-5f));
  }
}

// ---------------------------------------------------------------------------
// Tiled transpose + normalize: x[C,HW] f32 -> normT[HW,C] f16 and contT[HW,C]
// f16.  64x64 tiles through LDS; coalesced loads and stores.
// ---------------------------------------------------------------------------
__global__ __launch_bounds__(256)
void normT_kernel(const float* __restrict__ x, const float2* __restrict__ stats,
                  _Float16* __restrict__ normT, _Float16* __restrict__ contT) {
  __shared__ _Float16 ln[64 * 72];   // [spatial][channel], padded stride 72
  __shared__ _Float16 lc[64 * 72];
  const int t  = threadIdx.x;
  const int c0 = blockIdx.y * 64;
  const int i0 = blockIdx.x * 64;
#pragma unroll
  for (int ch = 0; ch < 4; ++ch) {
    const int row = (t >> 4) + ch * 16;   // channel-local
    const int col = (t & 15) * 4;         // spatial-local
    const float4 v = *(const float4*)(x + (size_t)(c0 + row) * HW_ + i0 + col);
    const float2 s = stats[c0 + row];
    const float vv[4] = {v.x, v.y, v.z, v.w};
#pragma unroll
    for (int j = 0; j < 4; ++j) {
      ln[(col + j) * 72 + row] = (_Float16)((vv[j] - s.x) * s.y);
      lc[(col + j) * 72 + row] = (_Float16)vv[j];
    }
  }
  __syncthreads();
  const int row = t >> 2;          // spatial-local 0..63
  const int col = (t & 3) * 16;    // channel-local
  _Float16* dn = normT + (size_t)(i0 + row) * C_ + c0 + col;
  _Float16* dc = contT + (size_t)(i0 + row) * C_ + c0 + col;
  const _Float16* sn = ln + row * 72 + col;
  const _Float16* sc = lc + row * 72 + col;
  *(uint4*)(dn)     = *(const uint4*)(sn);
  *(uint4*)(dn + 8) = *(const uint4*)(sn + 8);
  *(uint4*)(dc)     = *(const uint4*)(sc);
  *(uint4*)(dc + 8) = *(const uint4*)(sc + 8);
}

// ---------------------------------------------------------------------------
// f32 -> f16 elementwise (weight matrices)
// ---------------------------------------------------------------------------
__global__ __launch_bounds__(256)
void f32_to_f16_kernel(const float* __restrict__ src, _Float16* __restrict__ dst, int n) {
  int i = blockIdx.x * 256 + threadIdx.x;
  if (i < n) dst[i] = (_Float16)src[i];
}

// ---------------------------------------------------------------------------
// Row softmax: att[m, :] = softmax(energy[m, :]).  One block per row,
// row cached in registers (single global read, coalesced f16 write).
// ---------------------------------------------------------------------------
__global__ __launch_bounds__(256)
void softmax_kernel(const float* __restrict__ energy,  // [HW, HW]
                    _Float16* __restrict__ att) {      // [HW, HW] row-major
  __shared__ float red[256];
  const int m = blockIdx.x;
  const float* row = energy + (size_t)m * HW_;
  float v[16];
  float mx = -3.0e38f;
#pragma unroll
  for (int q = 0; q < 16; ++q) {
    v[q] = row[threadIdx.x + q * 256];
    mx = fmaxf(mx, v[q]);
  }
  red[threadIdx.x] = mx;
  __syncthreads();
  for (int st = 128; st > 0; st >>= 1) {
    if (threadIdx.x < st) red[threadIdx.x] = fmaxf(red[threadIdx.x], red[threadIdx.x + st]);
    __syncthreads();
  }
  mx = red[0];
  __syncthreads();
  float sum = 0.f;
#pragma unroll
  for (int q = 0; q < 16; ++q) {
    v[q] = __expf(v[q] - mx);
    sum += v[q];
  }
  red[threadIdx.x] = sum;
  __syncthreads();
  for (int st = 128; st > 0; st >>= 1) {
    if (threadIdx.x < st) red[threadIdx.x] += red[threadIdx.x + st];
    __syncthreads();
  }
  const float inv = 1.0f / red[0];
#pragma unroll
  for (int q = 0; q < 16; ++q)
    att[(size_t)m * HW_ + threadIdx.x + q * 256] = (_Float16)(v[q] * inv);
}

// ---------------------------------------------------------------------------
// TN WMMA GEMM:  C[M,N] = A[M,K] * Bt[N,K]^T   (A, Bt f16, K-contiguous)
// Fragments load DIRECTLY from global memory as 2x b128 each (no LDS staging):
//   A frag  (lane L): row M=m, K runs {8*(L/16)+0..7} and {16+8*(L/16)+0..7}
//   Bt frag (lane L): row N=n, K runs {8*(L/16)+0..7} and {16+8*(L/16)+0..7}
// 256 threads = 8 wave32; wave grid 2(M) x 4(N); wave tile 64(M) x 32(N)
// (4x2 fragments -> 8 WMMAs per K-step, 12 b128 loads); block tile 128x128.
// EPI: 0 = f32 store
//      2 = (+bias) f16 store
//      3 = (+bias) f16 TRANSPOSED store via LDS tile (out[n*M + m], coalesced)
//      4 = (+bias) + resid, f32 store
// ---------------------------------------------------------------------------
template <int EPI, bool HASB>
__global__ __launch_bounds__(256)
void gemm_tn_wmma_kernel(const _Float16* __restrict__ A,   // [M, K]
                         const _Float16* __restrict__ Bt,  // [N, K]
                         float* __restrict__ Cf,
                         _Float16* __restrict__ Ch,
                         const float* __restrict__ bias,
                         const float* __restrict__ resid,
                         int M, int N, int K) {
  __shared__ _Float16 lt[(EPI == 3) ? (128 * 136) : 4];

  const int t     = threadIdx.x;
  const int lane  = t & 31;
  const int w     = t >> 5;
  const int wm    = w & 1;        // 2 wave-rows of 64 in M
  const int wn    = w >> 1;       // 4 wave-cols of 32 in N
  const int lhalf = lane >> 4;
  const int l16   = lane & 15;
  const int m0    = blockIdx.y * 128;
  const int n0    = blockIdx.x * 128;

  const _Float16* Ar[4];
  const _Float16* Br[2];
#pragma unroll
  for (int as = 0; as < 4; ++as)
    Ar[as] = A + (size_t)(m0 + wm * 64 + as * 16 + l16) * K + lhalf * 8;
#pragma unroll
  for (int bs = 0; bs < 2; ++bs)
    Br[bs] = Bt + (size_t)(n0 + wn * 32 + bs * 16 + l16) * K + lhalf * 8;

  v8f acc[4][2] = {{{}, {}}, {{}, {}}, {{}, {}}, {{}, {}}};

  for (int k0 = 0; k0 < K; k0 += 32) {
    FragH a[4], bfr[2];
#pragma unroll
    for (int as = 0; as < 4; ++as) {
      a[as].q[0] = *(const uint4*)(Ar[as] + k0);
      a[as].q[1] = *(const uint4*)(Ar[as] + k0 + 16);
    }
#pragma unroll
    for (int bs = 0; bs < 2; ++bs) {
      bfr[bs].q[0] = *(const uint4*)(Br[bs] + k0);
      bfr[bs].q[1] = *(const uint4*)(Br[bs] + k0 + 16);
    }
#pragma unroll
    for (int as = 0; as < 4; ++as)
#pragma unroll
      for (int bs = 0; bs < 2; ++bs)
        acc[as][bs] = __builtin_amdgcn_wmma_f32_16x16x32_f16(
            false, a[as].v, false, bfr[bs].v, (short)0, acc[as][bs], false, false);
  }

  if (EPI == 3) {
    // Transposed f16 store, staged through LDS so global writes are coalesced.
#pragma unroll
    for (int as = 0; as < 4; ++as)
#pragma unroll
      for (int bs = 0; bs < 2; ++bs)
#pragma unroll
        for (int r = 0; r < 8; ++r) {
          const int ml = wm * 64 + as * 16 + r + (lhalf << 3);
          const int nl = wn * 32 + bs * 16 + l16;
          float v = acc[as][bs][r];
          if (HASB) v += bias[m0 + ml];
          lt[nl * 136 + ml] = (_Float16)v;
        }
    __syncthreads();
    const int row = t >> 1;          // n-local 0..127
    const int col = (t & 1) * 64;    // m-local
    _Float16* dst = Ch + (size_t)(n0 + row) * M + m0 + col;
    const _Float16* src = lt + row * 136 + col;
#pragma unroll
    for (int q = 0; q < 8; ++q)
      *(uint4*)(dst + q * 8) = *(const uint4*)(src + q * 8);
  } else {
#pragma unroll
    for (int as = 0; as < 4; ++as)
#pragma unroll
      for (int bs = 0; bs < 2; ++bs)
#pragma unroll
        for (int r = 0; r < 8; ++r) {
          const int m = m0 + wm * 64 + as * 16 + r + (lhalf << 3);
          const int n = n0 + wn * 32 + bs * 16 + l16;
          float v = acc[as][bs][r];
          if (HASB) v += bias[m];
          if (EPI == 0) {
            Cf[(size_t)m * N + n] = v;
          } else if (EPI == 2) {
            Ch[(size_t)m * N + n] = (_Float16)v;
          } else if (EPI == 4) {
            Cf[(size_t)m * N + n] = v + resid[(size_t)m * N + n];
          }
        }
  }
}

// ---------------------------------------------------------------------------
// Host orchestration
// ---------------------------------------------------------------------------
extern "C" void kernel_launch(void* const* d_in, const int* in_sizes, int n_in,
                              void* d_out, int out_size, void* d_ws, size_t ws_size,
                              hipStream_t stream) {
  (void)in_sizes; (void)n_in; (void)out_size; (void)ws_size;

  const float* content = (const float*)d_in[0];
  const float* f_w = (const float*)d_in[1];
  const float* f_b = (const float*)d_in[2];
  const float* g_w = (const float*)d_in[3];
  const float* g_b = (const float*)d_in[4];
  const float* h_w = (const float*)d_in[5];
  const float* h_b = (const float*)d_in[6];
  const float* o_w = (const float*)d_in[7];
  const float* o_b = (const float*)d_in[8];
  float* out = (float*)d_out;

  const int Cc = C_, HW = HW_, Bb = 4;

  char* p = (char*)d_ws;
  auto carve = [&](size_t bytes) -> char* {
    char* r = p;
    p += (bytes + 255) & ~(size_t)255;
    return r;
  };
  _Float16* fw16   = (_Float16*)carve((size_t)Cc * Cc * 2);
  _Float16* gw16   = (_Float16*)carve((size_t)Cc * Cc * 2);
  _Float16* hw16   = (_Float16*)carve((size_t)Cc * Cc * 2);
  _Float16* ow16   = (_Float16*)carve((size_t)Cc * Cc * 2);
  float2*   stats  = (float2*)  carve((size_t)Cc * sizeof(float2));
  _Float16* normT  = (_Float16*)carve((size_t)HW * Cc * 2);   // [HW, C]
  _Float16* contT  = (_Float16*)carve((size_t)HW * Cc * 2);   // [HW, C]
  _Float16* Ft16   = (_Float16*)carve((size_t)HW * Cc * 2);   // [HW, C]
  _Float16* Gt16   = (_Float16*)carve((size_t)HW * Cc * 2);   // [HW, C]
  _Float16* H16    = (_Float16*)carve((size_t)Cc * HW * 2);   // [C, HW]
  float*    energy = (float*)   carve((size_t)HW * HW * 4);   // [HW, HW] f32
  _Float16* att16  = (_Float16*)carve((size_t)HW * HW * 2);   // [HW, HW] row-major
  _Float16* aoT16  = (_Float16*)carve((size_t)HW * Cc * 2);   // [HW, C]

  // Convert weights to f16 once.
  {
    const int n = Cc * Cc;
    const int g = (n + 255) / 256;
    f32_to_f16_kernel<<<g, 256, 0, stream>>>(f_w, fw16, n);
    f32_to_f16_kernel<<<g, 256, 0, stream>>>(g_w, gw16, n);
    f32_to_f16_kernel<<<g, 256, 0, stream>>>(h_w, hw16, n);
    f32_to_f16_kernel<<<g, 256, 0, stream>>>(o_w, ow16, n);
  }

  const dim3 blk(256);
  const dim3 gridT(HW / 64, Cc / 64);        // transpose tiles
  const dim3 gridConv(HW / 128, Cc / 128);   // M=512,  N=4096
  const dim3 gridEng(HW / 128, HW / 128);    // M=4096, N=4096

  for (int b = 0; b < Bb; ++b) {
    const float* xb = content + (size_t)b * Cc * HW;
    float* ob = out + (size_t)b * Cc * HW;

    // 1) per-channel stats, then fused normalize + transpose to [HW, C]
    meanvar_kernel<<<Cc, blk, 0, stream>>>(xb, stats);
    normT_kernel<<<gridT, blk, 0, stream>>>(xb, stats, normT, contT);

    // 2) F^T = (f_w * norm)^T   [HW, C]
    gemm_tn_wmma_kernel<3, true><<<gridConv, blk, 0, stream>>>(
        fw16, normT, nullptr, Ft16, f_b, nullptr, Cc, HW, Cc);

    // 3) G^T = (g_w * norm)^T   [HW, C]
    gemm_tn_wmma_kernel<3, true><<<gridConv, blk, 0, stream>>>(
        gw16, normT, nullptr, Gt16, g_b, nullptr, Cc, HW, Cc);

    // 4) H = h_w * content      [C, HW]
    gemm_tn_wmma_kernel<2, true><<<gridConv, blk, 0, stream>>>(
        hw16, contT, nullptr, H16, h_b, nullptr, Cc, HW, Cc);

    // 5) energy[i,j] = sum_c Ft[i,c] * Gt[j,c]   [HW, HW] f32
    gemm_tn_wmma_kernel<0, false><<<gridEng, blk, 0, stream>>>(
        Ft16, Gt16, energy, nullptr, nullptr, nullptr, HW, HW, Cc);

    // 6) att = row-softmax(energy), natural row-major f16 (coalesced)
    softmax_kernel<<<HW, blk, 0, stream>>>(energy, att16);

    // 7) ao^T[m,c] = sum_n H[c,n] * att[m,n]   -> [HW, C]
    gemm_tn_wmma_kernel<3, false><<<gridConv, blk, 0, stream>>>(
        H16, att16, nullptr, aoT16, nullptr, nullptr, Cc, HW, HW);

    // 8) out = o_w * ao + o_b + content   (f32 -> d_out)
    gemm_tn_wmma_kernel<4, true><<<gridConv, blk, 0, stream>>>(
        ow16, aoT16, ob, nullptr, o_b, xb, Cc, HW, Cc);
  }
}
